// GNN_9594956939748
// MI455X (gfx1250) — matrix operand order repaired and checked by reading, not compiled, not run on hardware
//
#include <hip/hip_runtime.h>
#include <hip/hip_bf16.h>
#include <stdint.h>

// ---------------- problem constants ----------------
#define NN   40000      // nodes
#define EE   640000     // edges
#define RR   4          // relations
#define FF   128        // input features
#define HH   128        // hidden
#define NBLK 2          // conv blocks
#define GG   64         // graphs
#define NOUT 10         // outputs
#define NDEG 11         // degree buckets 0..10

typedef __attribute__((ext_vector_type(16))) __bf16 v16bf;
typedef __attribute__((ext_vector_type(8)))  float  v8f;
typedef int v4i __attribute__((vector_size(16)));
typedef unsigned int u32x4 __attribute__((ext_vector_type(4)));
typedef int i32x8 __attribute__((ext_vector_type(8)));
typedef int i32x4 __attribute__((ext_vector_type(4)));

// ---- CDNA5 feature probes (all guarded; fall back to plain paths) ----
#if defined(__gfx1250__) && __has_builtin(__builtin_amdgcn_global_load_async_to_lds_b128) && __has_builtin(__builtin_amdgcn_s_wait_asynccnt)
#define HAVE_ASYNC_LDS 1
#else
#define HAVE_ASYNC_LDS 0
#endif

#if defined(__gfx1250__) && __has_builtin(__builtin_amdgcn_tensor_load_to_lds) && __has_builtin(__builtin_amdgcn_s_wait_tensorcnt)
#define HAVE_TDM 1
#else
#define HAVE_TDM 0
#endif
// arity discriminator: therock headers (6-arg builtin) ship this header; ROCm 7.2 (5-arg) does not
#if __has_include(<hip/amd_detail/amd_gfx1250_TDM.h>)
#define TDM_6ARG 1
#else
#define TDM_6ARG 0
#endif

__device__ __forceinline__ void atomAddF(float* p, float v) {
#if defined(__HIP_DEVICE_COMPILE__)
  unsafeAtomicAdd(p, v);   // lowers to global_atomic_add_f32 on gfx1250
#else
  (void)p; (void)v;
#endif
}

__device__ __forceinline__ v8f wmma_bf16(v16bf a, v16bf b, v8f c) {
#if defined(__gfx1250__)
  return __builtin_amdgcn_wmma_f32_16x16x32_bf16(false, a, false, b, (short)0, c, false, false);
#else
  return c;
#endif
}

// per-lane async copy of one 16B chunk global->LDS
__device__ __forceinline__ void async_copy_b128(const void* gp, void* lp) {
#if HAVE_ASYNC_LDS
  __builtin_amdgcn_global_load_async_to_lds_b128(
      (__attribute__((address_space(1))) v4i*)gp,
      (__attribute__((address_space(3))) v4i*)lp, 0, 0);
#else
  *(uint4*)lp = *(const uint4*)gp;
#endif
}
__device__ __forceinline__ void async_wait_all() {
#if HAVE_ASYNC_LDS
  __builtin_amdgcn_s_wait_asynccnt(0);
#endif
}

#if HAVE_TDM
// TDM 2D tile load (2-byte elements): tile1 rows x tile0 elems, row stride stride0
// elems, from gsrc into contiguous LDS at ldst.  D# packing per CDNA5 ISA §8.3/8.4.
__device__ __forceinline__ void tdm_load_2d_b16(const void* gsrc, void* ldst,
                                                unsigned tile0, unsigned tile1,
                                                unsigned long long stride0,
                                                unsigned td0, unsigned td1) {
  unsigned long long ga = (unsigned long long)(size_t)gsrc;
  unsigned lds = (unsigned)(size_t)ldst;          // low 32 bits = LDS offset
  u32x4 g0 = {
      1u,                                         // count=1, user descriptor
      lds,                                        // lds_addr
      (unsigned)(ga & 0xffffffffu),               // global_addr[31:0]
      (unsigned)((ga >> 32) & 0x1ffffffu) | (2u << 30)  // global_addr[56:32] | type=2
  };
  i32x8 g1 = {
      (int)(1u << 16),                                            // data_size=1 (2B)
      (int)((td0 & 0xffffu) << 16),                               // tensor_dim0[15:0]
      (int)(((td0 >> 16) & 0xffffu) | ((td1 & 0xffffu) << 16)),   // td0 hi | td1 lo
      (int)(((td1 >> 16) & 0xffffu) | ((tile0 & 0xffffu) << 16)), // td1 hi | tile_dim0
      (int)(tile1 & 0xffffu),                                     // tile_dim1 (tile_dim2=0)
      (int)(unsigned)(stride0 & 0xffffffffu),                     // tensor_dim0_stride lo
      (int)(unsigned)((stride0 >> 32) & 0xffffu),                 // stride0 hi (stride1=0)
      0
  };
  i32x4 gz = {0, 0, 0, 0};
#if TDM_6ARG
  i32x8 gz8 = {0, 0, 0, 0, 0, 0, 0, 0};
  __builtin_amdgcn_tensor_load_to_lds(g0, g1, gz, gz, gz8, 0);
#else
  __builtin_amdgcn_tensor_load_to_lds(g0, g1, gz, gz, 0);
#endif
}
#endif

// ---------------- small utility kernels ----------------
__global__ void k_cast_bf16(const float* __restrict__ s, __bf16* __restrict__ d, int n) {
  int i = blockIdx.x * blockDim.x + threadIdx.x;
  if (i < n) d[i] = (__bf16)s[i];
}
__global__ void k_relu_cast(const float* __restrict__ s, __bf16* __restrict__ d, int n) {
  int i = blockIdx.x * blockDim.x + threadIdx.x;
  if (i < n) { float v = s[i]; v = v > 0.f ? v : 0.f; d[i] = (__bf16)v; }
}
__global__ void k_fill_f32(float* p, float v, int n) {
  int i = blockIdx.x * blockDim.x + threadIdx.x;
  if (i < n) p[i] = v;
}
__global__ void k_fill_i32(int* p, int v, int n) {
  int i = blockIdx.x * blockDim.x + threadIdx.x;
  if (i < n) p[i] = v;
}

// ---------------- graph preprocessing ----------------
__global__ void k_deg(const int* __restrict__ dst, int* __restrict__ deg) {
  int e = blockIdx.x * blockDim.x + threadIdx.x;
  if (e < EE) atomicAdd(&deg[dst[e]], 1);
}
__global__ void k_bucket_count(const int* __restrict__ deg, int* __restrict__ bcnt) {
  int n = blockIdx.x * blockDim.x + threadIdx.x;
  if (n < NN) { int d = deg[n]; d = d > NDEG - 1 ? NDEG - 1 : d; atomicAdd(&bcnt[d], 1); }
}
__global__ void k_bucket_prefix(const int* __restrict__ bcnt, int* __restrict__ boff, int* __restrict__ bpos) {
  if (blockIdx.x == 0 && threadIdx.x == 0) {
    int s = 0;
    for (int d = 0; d < NDEG; ++d) { boff[d] = s; bpos[d] = s; s += bcnt[d]; }
  }
}
__global__ void k_bucket_scatter(const int* __restrict__ deg, int* __restrict__ bpos, int* __restrict__ order) {
  int n = blockIdx.x * blockDim.x + threadIdx.x;
  if (n < NN) {
    int d = deg[n]; d = d > NDEG - 1 ? NDEG - 1 : d;
    int p = atomicAdd(&bpos[d], 1);
    order[p] = n;
  }
}

// ---------------- edge scatter kernels (memory-bound stage) ----------------
__global__ void k_rgcn_scatter(const __bf16* __restrict__ hb, const int* __restrict__ src,
                               const int* __restrict__ dst, const int* __restrict__ et,
                               float* __restrict__ rsum, float* __restrict__ rcnt) {
  int e = blockIdx.x, t = threadIdx.x;
  int s = src[e], d = dst[e], r = et[e];
  float v = (float)hb[(size_t)s * HH + t];
  atomAddF(&rsum[((size_t)d * RR + r) * HH + t], v);
  if (t == 0) atomAddF(&rcnt[d * RR + r], 1.0f);
}
__global__ void k_mf_scatter(const __bf16* __restrict__ hb, const int* __restrict__ src,
                             const int* __restrict__ dst, float* __restrict__ agg) {
  int e = blockIdx.x, t = threadIdx.x;
  int s = src[e], d = dst[e];
  atomAddF(&agg[(size_t)d * HH + t], (float)hb[(size_t)s * HH + t]);
}
__global__ void k_mean_cast(const float* __restrict__ rsum, const float* __restrict__ rcnt,
                            __bf16* __restrict__ mb) {
  int i = blockIdx.x * blockDim.x + threadIdx.x;
  if (i < NN * RR * HH) {
    float c = rcnt[i >> 7];
    float m = c > 0.f ? rsum[i] / c : 0.f;
    mb[i] = (__bf16)m;
  }
}

// ---------------- WMMA fragment helpers (contiguous k = 16*(lane/16) + s, same in A and B) ----------------
__device__ __forceinline__ v16bf load_a_frag(const __bf16* As /*row-major ld=32*/, int lane) {
  const __bf16* p = As + (size_t)(lane & 15) * 32 + ((lane & 16) ? 16 : 0);
  v16bf a;
#pragma unroll
  for (int i = 0; i < 8; ++i) a[i] = p[i];
#pragma unroll
  for (int i = 0; i < 8; ++i) a[8 + i] = p[8 + i];
  return a;
}
__device__ __forceinline__ v16bf load_b_frag(const __bf16* Bt /*n-major ld=32*/, int lane) {
  const __bf16* p = Bt + (size_t)(lane & 15) * 32 + ((lane & 16) ? 16 : 0);
  v16bf b;
#pragma unroll
  for (int i = 0; i < 16; ++i) b[i] = p[i];
  return b;
}

// ---------------- dense GEMM: C[M,128] = A[M,K] @ B[K,128] (+bias)(+accum) ----------------
// 256 threads = 8 waves; block tile 64x128; wave tile 16x64 (4 wmma accumulators).
// A tile staged by the Tensor Data Mover when available.
template <bool ACCUM, bool BIAS>
__launch_bounds__(256)
__global__ void k_gemm(const __bf16* __restrict__ A, const __bf16* __restrict__ B,
                       const float* __restrict__ bias, float* __restrict__ C, int K) {
  __shared__ __attribute__((aligned(16))) __bf16 As[64 * 32];
  __shared__ __attribute__((aligned(16))) __bf16 Bt[128 * 32];
  const int tid = threadIdx.x;
  const int lane = tid & 31, w = tid >> 5;
  const int wr = w & 3, wc = w >> 2;        // 4 row-waves x 2 col-waves
  const int row0 = blockIdx.x * 64;         // NN % 64 == 0
  const int rbase = (lane & 16) ? 8 : 0;
  const int col16 = lane & 15;

  v8f acc[4];
#pragma unroll
  for (int t = 0; t < 4; ++t) {
#pragma unroll
    for (int v = 0; v < 8; ++v) {
      if (ACCUM) {
        int row = row0 + wr * 16 + rbase + v;
        int col = wc * 64 + t * 16 + col16;
        acc[t][v] = C[(size_t)row * HH + col];
      } else {
        acc[t][v] = 0.f;
      }
    }
  }

  for (int k0 = 0; k0 < K; k0 += 32) {
#if HAVE_TDM
    if (tid < 32) {  // wave 0 drives the TDM; EXEC is ignored by tensor ops
      tdm_load_2d_b16(A + (size_t)row0 * K + k0, As,
                      /*tile0=*/32u, /*tile1=*/64u,
                      /*stride0=*/(unsigned long long)K,
                      /*td0=*/(unsigned)K, /*td1=*/64u);
      __builtin_amdgcn_s_wait_tensorcnt(0);
    }
#else
    { // per-lane 16B copies (async when available)
      int r = tid >> 2, cb = (tid & 3) * 8;
      const __bf16* gp = A + (size_t)(row0 + r) * K + k0 + cb;
      async_copy_b128(gp, &As[r * 32 + cb]);
      if (k0 + 32 < K) __builtin_prefetch(gp + 32, 0, 3);
    }
#endif
    { // stage B tile 32x128, transposed into Bt[n][k]
      int kk = tid >> 3, nb = (tid & 7) * 16;
      const __bf16* gp = B + (size_t)(k0 + kk) * HH + nb;
#pragma unroll
      for (int j = 0; j < 16; ++j) Bt[(nb + j) * 32 + kk] = gp[j];
    }
#if !HAVE_TDM
    async_wait_all();
#endif
    __syncthreads();

    v16bf a = load_a_frag(&As[(wr * 16) * 32], lane);
#pragma unroll
    for (int t = 0; t < 4; ++t) {
      v16bf b = load_b_frag(&Bt[(wc * 64 + t * 16) * 32], lane);
      acc[t] = wmma_bf16(a, b, acc[t]);
    }
    __syncthreads();
  }

#pragma unroll
  for (int t = 0; t < 4; ++t) {
#pragma unroll
    for (int v = 0; v < 8; ++v) {
      int row = row0 + wr * 16 + rbase + v;
      int col = wc * 64 + t * 16 + col16;
      float x = acc[t][v];
      if (BIAS) x += bias[col];
      C[(size_t)row * HH + col] = x;
    }
  }
}

// ---------------- MFConv bucketed/gathered GEMM ----------------
// out[node] = agg[node]@Wl[d] + bl[d] + h[node]@Wr[d]; nodes grouped by degree bucket.
// K=256 GEMM with A=[agg|h], B=[Wl[d];Wr[d]].  Gathered A rows staged with per-lane
// async global->LDS loads (tail rows clamp to node 0, masked at the store).
__launch_bounds__(256)
__global__ void k_mf_gemm(const __bf16* __restrict__ Aagg, const __bf16* __restrict__ Ah,
                          const __bf16* __restrict__ Wl, const __bf16* __restrict__ Wr,
                          const float* __restrict__ bl,
                          const int* __restrict__ order, const int* __restrict__ boff,
                          const int* __restrict__ bcnt, float* __restrict__ C) {
  const int d = blockIdx.y;
  const int cnt = bcnt[d], base = boff[d];
  const int row0 = blockIdx.x * 64;
  if (row0 >= cnt) return;                         // block-uniform exit

  __shared__ __attribute__((aligned(16))) __bf16 As[64 * 32];
  __shared__ __attribute__((aligned(16))) __bf16 Bt[128 * 32];
  __shared__ int nodes[64];

  const int tid = threadIdx.x;
  const int lane = tid & 31, w = tid >> 5;
  const int wr = w & 3, wc = w >> 2;
  const int rbase = (lane & 16) ? 8 : 0;
  const int col16 = lane & 15;

  if (tid < 64) {
    int pos = row0 + tid;
    nodes[tid] = (pos < cnt) ? order[base + pos] : -1;
  }
  __syncthreads();

  const __bf16* Bl = Wl + (size_t)d * HH * HH;
  const __bf16* Br = Wr + (size_t)d * HH * HH;

  v8f acc[4];
#pragma unroll
  for (int t = 0; t < 4; ++t)
#pragma unroll
    for (int v = 0; v < 8; ++v) acc[t][v] = 0.f;

  for (int k0 = 0; k0 < 256; k0 += 32) {
    const int part = k0 >> 7, kloc = k0 & 127;
    const __bf16* Ap = part ? Ah : Aagg;
    const __bf16* Bp = part ? Br : Bl;
    { // gathered A tile: per-lane async copy, invalid rows clamped to node 0
      int r = tid >> 2, cb = (tid & 3) * 8;
      int node = nodes[r];
      int sn = node < 0 ? 0 : node;
      async_copy_b128(Ap + (size_t)sn * HH + kloc + cb, &As[r * 32 + cb]);
    }
    { // B tile transposed
      int kk = tid >> 3, nb = (tid & 7) * 16;
      const __bf16* gp = Bp + (size_t)(kloc + kk) * HH + nb;
#pragma unroll
      for (int j = 0; j < 16; ++j) Bt[(nb + j) * 32 + kk] = gp[j];
    }
    async_wait_all();
    __syncthreads();

    v16bf a = load_a_frag(&As[(wr * 16) * 32], lane);
#pragma unroll
    for (int t = 0; t < 4; ++t) {
      v16bf b = load_b_frag(&Bt[(wc * 64 + t * 16) * 32], lane);
      acc[t] = wmma_bf16(a, b, acc[t]);
    }
    __syncthreads();
  }

  const float* biasd = bl + d * HH;
#pragma unroll
  for (int t = 0; t < 4; ++t) {
#pragma unroll
    for (int v = 0; v < 8; ++v) {
      int rl = wr * 16 + rbase + v;
      int node = nodes[rl];
      if (node >= 0) {
        int col = wc * 64 + t * 16 + col16;
        C[(size_t)node * HH + col] = acc[t][v] + biasd[col];
      }
    }
  }
}

// ---------------- pooling + output MLP ----------------
__global__ void k_pool(const float* __restrict__ h, const int* __restrict__ bidx,
                       float* __restrict__ pooled) {
  int n = blockIdx.x, t = threadIdx.x;
  atomAddF(&pooled[(size_t)bidx[n] * HH + t], h[(size_t)n * HH + t]);
}
__global__ void k_mlp(const float* __restrict__ pooled, const float* __restrict__ W1,
                      const float* __restrict__ b1, const float* __restrict__ W2,
                      const float* __restrict__ b2, float* __restrict__ out) {
  __shared__ float pr[HH];
  __shared__ float hid[HH];
  int g = blockIdx.x, t = threadIdx.x;
  pr[t] = pooled[(size_t)g * HH + t];
  __syncthreads();
  float s = b1[t];
  for (int k = 0; k < HH; ++k) s += pr[k] * W1[k * HH + t];
  hid[t] = s > 0.f ? s : 0.f;
  __syncthreads();
  if (t < NOUT) {
    float o = b2[t];
    for (int k = 0; k < HH; ++k) o += hid[k] * W2[k * NOUT + t];
    out[(size_t)g * NOUT + t] = o;
  }
}

// ---------------- host orchestration ----------------
extern "C" void kernel_launch(void* const* d_in, const int* in_sizes, int n_in,
                              void* d_out, int out_size, void* d_ws, size_t ws_size,
                              hipStream_t stream) {
  (void)in_sizes; (void)n_in; (void)out_size; (void)ws_size;
  const float* x     = (const float*)d_in[0];
  const int*   eidx  = (const int*)  d_in[1];
  const int*   eattr = (const int*)  d_in[2];
  const int*   bidx  = (const int*)  d_in[3];
  const float* W0    = (const float*)d_in[4];
  const float* b0    = (const float*)d_in[5];
  const float* Wrel  = (const float*)d_in[6];
  const float* Wroot = (const float*)d_in[7];
  const float* rgb   = (const float*)d_in[8];
  const float* mfWl  = (const float*)d_in[9];
  const float* mfbl  = (const float*)d_in[10];
  const float* mfWr  = (const float*)d_in[11];
  const float* W1    = (const float*)d_in[12];
  const float* b1    = (const float*)d_in[13];
  const float* W2    = (const float*)d_in[14];
  const float* b2    = (const float*)d_in[15];
  float* out = (float*)d_out;
  const int* src = eidx;
  const int* dst = eidx + EE;

  // workspace carve-out (all sizes static; ~220 MB total)
  char* ws = (char*)d_ws;
  size_t off = 0;
  auto carve = [&](size_t bytes) -> char* {
    char* p = ws + off;
    off += (bytes + 255) & ~(size_t)255;
    return p;
  };
  float*  hA     = (float*)carve((size_t)NN * HH * 4);
  float*  hB     = (float*)carve((size_t)NN * HH * 4);
  float*  rsum   = (float*)carve((size_t)NN * RR * HH * 4);
  float*  rcnt   = (float*)carve((size_t)NN * RR * 4);
  float*  agg    = (float*)carve((size_t)NN * HH * 4);
  float*  pooled = (float*)carve((size_t)GG * HH * 4);
  int*    deg    = (int*)  carve((size_t)NN * 4);
  int*    bcnt   = (int*)  carve(NDEG * 4);
  int*    boff   = (int*)  carve(NDEG * 4);
  int*    bpos   = (int*)  carve(NDEG * 4);
  int*    order  = (int*)  carve((size_t)NN * 4);
  __bf16* xb     = (__bf16*)carve((size_t)NN * FF * 2);
  __bf16* hb     = (__bf16*)carve((size_t)NN * HH * 2);
  __bf16* aggb   = (__bf16*)carve((size_t)NN * HH * 2);
  __bf16* meanb  = (__bf16*)carve((size_t)NN * RR * HH * 2);
  __bf16* W0b    = (__bf16*)carve((size_t)FF * HH * 2);
  __bf16* Wrelb  = (__bf16*)carve((size_t)NBLK * RR * HH * HH * 2);
  __bf16* Wrootb = (__bf16*)carve((size_t)NBLK * HH * HH * 2);
  __bf16* Wlb    = (__bf16*)carve((size_t)NBLK * NDEG * HH * HH * 2);
  __bf16* Wrb    = (__bf16*)carve((size_t)NBLK * NDEG * HH * HH * 2);

  auto castN = [&](const float* s, __bf16* d, int n) {
    k_cast_bf16<<<(n + 255) / 256, 256, 0, stream>>>(s, d, n);
  };

  // ---- weight / input conversion to bf16 ----
  castN(x, xb, NN * FF);
  castN(W0, W0b, FF * HH);
  castN(Wrel, Wrelb, NBLK * RR * HH * HH);
  castN(Wroot, Wrootb, NBLK * HH * HH);
  castN(mfWl, Wlb, NBLK * NDEG * HH * HH);
  castN(mfWr, Wrb, NBLK * NDEG * HH * HH);

  // ---- degrees + degree buckets (constant across blocks) ----
  k_fill_i32<<<(NN + 255) / 256, 256, 0, stream>>>(deg, 0, NN);
  k_deg<<<(EE + 255) / 256, 256, 0, stream>>>(dst, deg);
  k_fill_i32<<<1, 64, 0, stream>>>(bcnt, 0, NDEG);
  k_bucket_count<<<(NN + 255) / 256, 256, 0, stream>>>(deg, bcnt);
  k_bucket_prefix<<<1, 1, 0, stream>>>(bcnt, boff, bpos);
  k_bucket_scatter<<<(NN + 255) / 256, 256, 0, stream>>>(deg, bpos, order);

  // ---- node embedding: hA = x @ W0 + b0 ----
  k_gemm<false, true><<<NN / 64, 256, 0, stream>>>(xb, W0b, b0, hA, FF);

  float* cur = hA;
  float* nxt = hB;
  const int tilesMF = (NN + 63) / 64;

  for (int blk = 0; blk < NBLK; ++blk) {
    // h = relu(h) -> bf16
    k_relu_cast<<<(NN * HH + 255) / 256, 256, 0, stream>>>(cur, hb, NN * HH);

    // RGCN: per-(dst,rel) mean of neighbors
    k_fill_f32<<<(NN * RR * HH + 255) / 256, 256, 0, stream>>>(rsum, 0.f, NN * RR * HH);
    k_fill_f32<<<(NN * RR + 255) / 256, 256, 0, stream>>>(rcnt, 0.f, NN * RR);
    k_rgcn_scatter<<<EE, HH, 0, stream>>>(hb, src, dst, eattr, rsum, rcnt);
    k_mean_cast<<<(NN * RR * HH + 255) / 256, 256, 0, stream>>>(rsum, rcnt, meanb);

    // nxt = mean[N, R*H] @ Wrel[R*H, H] + b ; then accumulate root term hb @ Wroot
    k_gemm<false, true><<<NN / 64, 256, 0, stream>>>(
        meanb, Wrelb + (size_t)blk * RR * HH * HH, rgb + blk * HH, nxt, RR * HH);
    k_gemm<true, false><<<NN / 64, 256, 0, stream>>>(
        hb, Wrootb + (size_t)blk * HH * HH, nullptr, nxt, HH);
    { float* t = cur; cur = nxt; nxt = t; }

    // MFConv: h = relu(h); agg = scatter-sum(h[src] -> dst)
    k_relu_cast<<<(NN * HH + 255) / 256, 256, 0, stream>>>(cur, hb, NN * HH);
    k_fill_f32<<<(NN * HH + 255) / 256, 256, 0, stream>>>(agg, 0.f, NN * HH);
    k_mf_scatter<<<EE, HH, 0, stream>>>(hb, src, dst, agg);
    castN(agg, aggb, NN * HH);

    // degree-bucketed GEMM: nxt[node] = agg@Wl[d] + bl[d] + h@Wr[d]
    dim3 gmf(tilesMF, NDEG);
    k_mf_gemm<<<gmf, 256, 0, stream>>>(aggb, hb,
                                       Wlb + (size_t)blk * NDEG * HH * HH,
                                       Wrb + (size_t)blk * NDEG * HH * HH,
                                       mfbl + (size_t)blk * NDEG * HH,
                                       order, boff, bcnt, nxt);
    { float* t = cur; cur = nxt; nxt = t; }
  }

  // ---- global add pool + output MLP ----
  k_fill_f32<<<(GG * HH + 255) / 256, 256, 0, stream>>>(pooled, 0.f, GG * HH);
  k_pool<<<NN, HH, 0, stream>>>(cur, bidx, pooled);
  k_mlp<<<GG, HH, 0, stream>>>(pooled, W1, b1, W2, b2, out);
}
